// TextCNNV2Attn_22582938042436
// MI455X (gfx1250) — matrix operand order repaired and checked by reading, not compiled, not run on hardware
//
#include <hip/hip_runtime.h>

// ---------------------------------------------------------------------------
// CDNA5 (gfx1250) implementation of the VQ-VAE text model forward pass.
// All dense math goes through v_wmma_f32_16x16x32_bf16 (fp32 -> bf16 inputs,
// f32 accumulate), tiles staged in LDS (64x128x64 per workgroup, 8 waves,
// wave32, 8 WMMA per barrier pair). Bias/activation/alpha fused in epilogue.
// Edge tiles use clamped loads + selects (no per-element exec branching).
// ---------------------------------------------------------------------------

typedef __attribute__((ext_vector_type(16))) __bf16 v16bf;
typedef __attribute__((ext_vector_type(8)))  __bf16 bf16x8;
typedef __attribute__((ext_vector_type(8)))  float  v8f;

#define TBM 64
#define TBN 128
#define TBK 64
#define LDT 72   // padded LDS row stride (bf16) -> all fragment chunks 16B aligned

// ---------------------------------------------------------------------------
// Generic batched GEMM:  C = act(alpha * A @ op(B) + bias)
//  A: [M,K] row-major (lda), per-batch offset sA
//  B: transB ? [N,K] (ldb) : [K,N] (ldb), per-batch offset sB
//  C: transC ? C[n*ldc+m] : C[m*ldc+n], per-batch offset sC
//  biasMode: 0 none, 1 along N, 2 along M ; act: 0 none, 1 relu, 2 elu
// ---------------------------------------------------------------------------
__global__ __launch_bounds__(256) void k_gemm(
    const float* __restrict__ A, const float* __restrict__ Bm,
    const float* __restrict__ bias, float* __restrict__ C,
    int M, int N, int K, int lda, int ldb, int ldc,
    long sA, long sB, long sC,
    float alpha, int transB, int transC, int biasMode, int act)
{
  __shared__ __attribute__((aligned(16))) __bf16 As[TBM * LDT];
  __shared__ __attribute__((aligned(16))) __bf16 Bs[TBN * LDT];

  const int tid = threadIdx.x;
  const long bz = blockIdx.z;
  A  += bz * sA;
  Bm += bz * sB;
  C  += bz * sC;

  const int m0 = blockIdx.y * TBM;
  const int n0 = blockIdx.x * TBN;
  const int wave = tid >> 5, lane = tid & 31;
  const int wm = (wave >> 2) * 32;       // 2 wave-rows
  const int wn = (wave & 3) * 32;        // 4 wave-cols
  const int half = lane >> 4, lr = lane & 15;

  v8f acc[2][2] = {};

  for (int k0 = 0; k0 < K; k0 += TBK) {
    const bool tailK = (k0 + TBK > K);
    // ---- stage A tile (64x64) : 16 elems / thread ----
    {
      int r = tid >> 2, kk = (tid & 3) * 16;
      int gm = m0 + r, gk = k0 + kk;
      __bf16* dst = &As[r * LDT + kk];
      if ((m0 + TBM <= M) && !tailK) {
        const float4* p = (const float4*)(A + (long)gm * lda + gk);
#pragma unroll
        for (int q4 = 0; q4 < 4; q4++) {
          float4 u = p[q4];
          dst[q4 * 4 + 0] = (__bf16)u.x; dst[q4 * 4 + 1] = (__bf16)u.y;
          dst[q4 * 4 + 2] = (__bf16)u.z; dst[q4 * 4 + 3] = (__bf16)u.w;
        }
        if (k0 + TBK < K) __builtin_prefetch(A + (long)gm * lda + gk + TBK, 0, 1);
      } else {
        // clamped addresses + value select: no exec-mask churn on edge tiles
        int gmc = gm < M ? gm : (M - 1);
        const float* src = A + (long)gmc * lda;
        bool rowOK = (gm < M);
#pragma unroll
        for (int i = 0; i < 16; i++) {
          int kg = gk + i;
          int kgc = kg < K ? kg : (K - 1);
          float v = src[kgc];
          dst[i] = (__bf16)((rowOK && kg < K) ? v : 0.f);
        }
      }
    }
    // ---- stage B tile (128x64, stored [N][K]) ----
    if (transB) {            // B is [N,K]: 32 contiguous K per thread
      int r = tid >> 1, kk = (tid & 1) * 32;
      int gn = n0 + r, gk = k0 + kk;
      __bf16* dst = &Bs[r * LDT + kk];
      if ((n0 + TBN <= N) && !tailK) {
        const float4* p = (const float4*)(Bm + (long)gn * ldb + gk);
#pragma unroll
        for (int q4 = 0; q4 < 8; q4++) {
          float4 u = p[q4];
          dst[q4 * 4 + 0] = (__bf16)u.x; dst[q4 * 4 + 1] = (__bf16)u.y;
          dst[q4 * 4 + 2] = (__bf16)u.z; dst[q4 * 4 + 3] = (__bf16)u.w;
        }
        if (k0 + TBK < K) __builtin_prefetch(Bm + (long)gn * ldb + gk + TBK, 0, 1);
      } else {
        int gnc = gn < N ? gn : (N - 1);
        const float* src = Bm + (long)gnc * ldb;
        bool rowOK = (gn < N);
#pragma unroll
        for (int i = 0; i < 32; i++) {
          int kg = gk + i;
          int kgc = kg < K ? kg : (K - 1);
          float v = src[kgc];
          dst[i] = (__bf16)((rowOK && kg < K) ? v : 0.f);
        }
      }
    } else {                 // B is [K,N]: read 32 contiguous N of one K-row
      int kloc = tid >> 2, ng = (tid & 3) * 32;
      int gk = k0 + kloc, gn0 = n0 + ng;
      if ((n0 + TBN <= N) && !tailK) {
        const float4* p = (const float4*)(Bm + (long)gk * ldb + gn0);
#pragma unroll
        for (int q4 = 0; q4 < 8; q4++) {
          float4 u = p[q4];
          Bs[(ng + q4 * 4 + 0) * LDT + kloc] = (__bf16)u.x;
          Bs[(ng + q4 * 4 + 1) * LDT + kloc] = (__bf16)u.y;
          Bs[(ng + q4 * 4 + 2) * LDT + kloc] = (__bf16)u.z;
          Bs[(ng + q4 * 4 + 3) * LDT + kloc] = (__bf16)u.w;
        }
        if (k0 + TBK < K) __builtin_prefetch(Bm + (long)(gk + TBK) * ldb + gn0, 0, 1);
      } else {
        int gkc = gk < K ? gk : (K - 1);
        const float* src = Bm + (long)gkc * ldb;
        bool rowOK = (gk < K);
#pragma unroll
        for (int i = 0; i < 32; i++) {
          int gn = gn0 + i;
          int gnc = gn < N ? gn : (N - 1);
          float v = src[gnc];
          Bs[(ng + i) * LDT + kloc] = (__bf16)((rowOK && gn < N) ? v : 0.f);
        }
      }
    }
    __syncthreads();

    // ---- 2 x (fragments + 4 WMMA): 8 v_wmma per barrier pair ----
#pragma unroll
    for (int kc = 0; kc < 2; kc++) {
      v16bf afr[2], bfr[2];
#pragma unroll
      for (int mi = 0; mi < 2; mi++) {
        // A lane layout: row=lr, K chunks {half*8..+8} and {16+half*8..+8}
        const __bf16* ap = &As[(wm + mi * 16 + lr) * LDT + kc * 32 + half * 8];
        bf16x8 lo = *(const bf16x8*)ap;
        bf16x8 hi = *(const bf16x8*)(ap + 16);
        afr[mi] = __builtin_shufflevector(lo, hi, 0,1,2,3,4,5,6,7,8,9,10,11,12,13,14,15);
      }
#pragma unroll
      for (int ni = 0; ni < 2; ni++) {
        // B lane layout: col=lr, K chunks {half*8..+8} and {16+half*8..+8}
        const __bf16* bp = &Bs[(wn + ni * 16 + lr) * LDT + kc * 32 + half * 8];
        bf16x8 lo = *(const bf16x8*)bp;
        bf16x8 hi = *(const bf16x8*)(bp + 16);
        bfr[ni] = __builtin_shufflevector(lo, hi, 0,1,2,3,4,5,6,7,8,9,10,11,12,13,14,15);
      }
#pragma unroll
      for (int mi = 0; mi < 2; mi++)
#pragma unroll
        for (int ni = 0; ni < 2; ni++)
          acc[mi][ni] = __builtin_amdgcn_wmma_f32_16x16x32_bf16(
              false, afr[mi], false, bfr[ni], (short)0, acc[mi][ni], false, false);
    }
    __syncthreads();
  }

  // ---- epilogue: alpha, bias, activation, (optionally transposed) store ----
#pragma unroll
  for (int mi = 0; mi < 2; mi++) {
#pragma unroll
    for (int ni = 0; ni < 2; ni++) {
#pragma unroll
      for (int r = 0; r < 8; r++) {
        int gm = m0 + wm + mi * 16 + half * 8 + r;
        int gn = n0 + wn + ni * 16 + lr;
        if (gm < M && gn < N) {
          float v = acc[mi][ni][r] * alpha;
          if (biasMode == 1) v += bias[gn];
          else if (biasMode == 2) v += bias[gm];
          if (act == 1) v = fmaxf(v, 0.f);
          else if (act == 2) v = v > 0.f ? v : (__expf(v) - 1.f);
          C[transC ? ((long)gn * ldc + gm) : ((long)gm * ldc + gn)] = v;
        }
      }
    }
  }
}

// ---------------------------------------------------------------------------
// Support kernels
// ---------------------------------------------------------------------------
__device__ __forceinline__ float pe_val(int pos, int ch) {
  int j = ch >> 1;
  float div = __expf(-(float)(2 * j) * 0.035977892944f);  // ln(10000)/256
  float a = (float)pos * div;
  return (ch & 1) ? cosf(a) : sinf(a);
}

// encoder embedding: renorm(max_norm=1) * sqrt(256) + PE, write [B,256,L]
__global__ __launch_bounds__(256) void k_embed_enc(const int* __restrict__ x,
                                                   const float* __restrict__ emb,
                                                   float* __restrict__ out) {
  int bid = blockIdx.x, l = bid % 512, b = bid / 512, tid = threadIdx.x;
  int tok = x[b * 512 + l];
  float w = emb[(long)tok * 256 + tid];
  __shared__ float red[256];
  red[tid] = w * w; __syncthreads();
  for (int st = 128; st > 0; st >>= 1) { if (tid < st) red[tid] += red[tid + st]; __syncthreads(); }
  float nrm = sqrtf(red[0]);
  float scale = fminf(1.f, 1.f / fmaxf(nrm, 1e-12f));
  out[((long)b * 256 + tid) * 512 + l] = w * scale * 16.f + pe_val(l, tid);
}

// decoder embedding: BOS-shifted tokens * sqrt(256) + PE, write [L,B,256]
__global__ __launch_bounds__(256) void k_embed_dec(const int* __restrict__ x,
                                                   const float* __restrict__ emb,
                                                   float* __restrict__ out) {
  int bid = blockIdx.x, b = bid % 16, l = bid / 16, tid = threadIdx.x;
  int tok = (l == 0) ? 10000 : x[b * 512 + (l - 1)];
  out[((long)l * 16 + b) * 256 + tid] =
      emb[(long)tok * 256 + tid] * 16.f + pe_val(l, tid);
}

__global__ __launch_bounds__(256) void k_elu(float* __restrict__ out,
                                             const float* __restrict__ in, long n) {
  long i = (long)blockIdx.x * 256 + threadIdx.x;
  if (i >= n) return;
  float v = in[i];
  out[i] = v > 0.f ? v : (__expf(v) - 1.f);
}

__global__ __launch_bounds__(256) void k_im2col3(const float* __restrict__ x,
                                                 float* __restrict__ col,
                                                 int C, int T, long n) {
  long i = (long)blockIdx.x * 256 + threadIdx.x;
  if (i >= n) return;
  int t = (int)(i % T); long r = i / T;
  int dk = (int)(r % 3); r /= 3;
  int c = (int)(r % C); int b = (int)(r / C);
  int ti = t + dk - 1;
  col[i] = (ti >= 0 && ti < T) ? x[((long)(b * C + c)) * T + ti] : 0.f;
}

__global__ __launch_bounds__(256) void k_im2col4(const float* __restrict__ x,
                                                 float* __restrict__ col,
                                                 int C, int T, int T2, long n) {
  long i = (long)blockIdx.x * 256 + threadIdx.x;
  if (i >= n) return;
  int t2 = (int)(i % T2); long r = i / T2;
  int dk = (int)(r % 4); r /= 4;
  int c = (int)(r % C); int b = (int)(r / C);
  int ti = 2 * t2 + dk - 1;
  col[i] = (ti >= 0 && ti < T) ? x[((long)(b * C + c)) * T + ti] : 0.f;
}

// channel layernorm over C=256 of [B,256,T], out = LN(hnew + res)
__global__ __launch_bounds__(256) void k_cln(const float* __restrict__ hnew,
                                             const float* __restrict__ res,
                                             float* __restrict__ out,
                                             const float* __restrict__ g,
                                             const float* __restrict__ bb,
                                             int T, int total) {
  int i = blockIdx.x * 256 + threadIdx.x;
  if (i >= total) return;
  int b = i / T, t = i % T;
  long base = (long)b * 256 * T + t;
  float mu = 0.f;
  for (int c = 0; c < 256; c++) mu += hnew[base + (long)c * T] + res[base + (long)c * T];
  mu *= (1.f / 256.f);
  float var = 0.f;
  for (int c = 0; c < 256; c++) {
    float v = hnew[base + (long)c * T] + res[base + (long)c * T] - mu;
    var += v * v;
  }
  var *= (1.f / 256.f);
  float inv = rsqrtf(var + 1e-5f);
  for (int c = 0; c < 256; c++) {
    float v = hnew[base + (long)c * T] + res[base + (long)c * T];
    out[base + (long)c * T] = (v - mu) * inv * g[c] + bb[c];
  }
}

// layernorm over D=256 rows: out = LN(x + res)
__global__ __launch_bounds__(256) void k_ln(float* __restrict__ out,
                                            const float* __restrict__ x,
                                            const float* __restrict__ res,
                                            const float* __restrict__ g,
                                            const float* __restrict__ bb) {
  long row = blockIdx.x; int tid = threadIdx.x;
  float v = x[row * 256 + tid];
  if (res) v += res[row * 256 + tid];
  __shared__ float red[256];
  red[tid] = v; __syncthreads();
  for (int st = 128; st > 0; st >>= 1) { if (tid < st) red[tid] += red[tid + st]; __syncthreads(); }
  float mu = red[0] * (1.f / 256.f); __syncthreads();
  float d = v - mu;
  red[tid] = d * d; __syncthreads();
  for (int st = 128; st > 0; st >>= 1) { if (tid < st) red[tid] += red[tid + st]; __syncthreads(); }
  float var = red[0] * (1.f / 256.f);
  out[row * 256 + tid] = d * rsqrtf(var + 1e-5f) * g[tid] + bb[tid];
}

// row softmax (len T), causal: only t <= (row % S) participates
__global__ __launch_bounds__(256) void k_softmax(float* __restrict__ s,
                                                 int T, int S, int causal) {
  long row = blockIdx.x; int tid = threadIdx.x;
  float* p = s + row * T;
  int lim = causal ? (int)(row % S) : (T - 1);
  __shared__ float red[256];
  float m = -3.4e38f;
  for (int t = tid; t < T; t += 256)
    if (t <= lim) { float v = p[t]; m = v > m ? v : m; }
  red[tid] = m; __syncthreads();
  for (int st = 128; st > 0; st >>= 1) { if (tid < st) red[tid] = fmaxf(red[tid], red[tid + st]); __syncthreads(); }
  m = red[0]; __syncthreads();
  float sum = 0.f;
  for (int t = tid; t < T; t += 256) {
    float e = (t <= lim) ? __expf(p[t] - m) : 0.f;
    p[t] = e; sum += e;
  }
  red[tid] = sum; __syncthreads();
  for (int st = 128; st > 0; st >>= 1) { if (tid < st) red[tid] += red[tid + st]; __syncthreads(); }
  float inv = 1.f / red[0];
  for (int t = tid; t < T; t += 256) p[t] *= inv;
}

// [B,256,T] -> [T,B,256]
__global__ __launch_bounds__(256) void k_bct_to_tbc(const float* __restrict__ in,
                                                    float* __restrict__ out, int T, long n) {
  long i = (long)blockIdx.x * 256 + threadIdx.x;
  if (i >= n) return;
  int t = (int)(i % T); long r = i / T;
  int c = (int)(r % 256); int b = (int)(r / 256);
  out[((long)t * 16 + b) * 256 + c] = in[i];
}

// out[B,256,T] = tbc[T,B,256] + skip[B,256,T]
__global__ __launch_bounds__(256) void k_tbc_add_skip(const float* __restrict__ tbc,
                                                      const float* __restrict__ skip,
                                                      float* __restrict__ out, int T, long n) {
  long i = (long)blockIdx.x * 256 + threadIdx.x;
  if (i >= n) return;
  int t = (int)(i % T); long r = i / T;
  int c = (int)(r % 256); int b = (int)(r / 256);
  out[i] = skip[i] + tbc[((long)t * 16 + b) * 256 + c];
}

// VQ: argmin over 512 codes of |x - e_j|^2 ; q = code ; diff += mean(q-x)^2
__global__ __launch_bounds__(256) void k_quantize(const float* __restrict__ xq,
                                                  const float* __restrict__ codes,  // [64,512]
                                                  float* __restrict__ qout,
                                                  float* __restrict__ iout,
                                                  float* __restrict__ diff, float dscale) {
  int row = blockIdx.x, tid = threadIdx.x;
  __shared__ float xs[64];
  __shared__ float bval[256];
  __shared__ int   bidx[256];
  __shared__ float dred[64];
  if (tid < 64) xs[tid] = xq[(long)row * 64 + tid];
  __syncthreads();
  float best = 3.4e38f; int bj = 0;
  for (int j = tid; j < 512; j += 256) {
    float s = 0.f;
#pragma unroll 4
    for (int d = 0; d < 64; d++) { float dd = xs[d] - codes[d * 512 + j]; s += dd * dd; }
    if (s < best) { best = s; bj = j; }
  }
  bval[tid] = best; bidx[tid] = bj; __syncthreads();
  for (int st = 128; st > 0; st >>= 1) {
    if (tid < st) {
      if (bval[tid + st] < bval[tid] ||
          (bval[tid + st] == bval[tid] && bidx[tid + st] < bidx[tid])) {
        bval[tid] = bval[tid + st]; bidx[tid] = bidx[tid + st];
      }
    }
    __syncthreads();
  }
  int jm = bidx[0];
  if (tid < 64) {
    float qv = codes[tid * 512 + jm];
    qout[(long)row * 64 + tid] = qv;
    float dd = qv - xs[tid];
    dred[tid] = dd * dd;
  }
  __syncthreads();
  if (tid == 0) {
    float s = 0.f;
    for (int d = 0; d < 64; d++) s += dred[d];
    atomicAdd(diff, s * dscale);
    iout[row] = (float)jm;
  }
}

// log_softmax over 10001, [L,B,:] -> out [B,L,:]
__global__ __launch_bounds__(256) void k_logsoftmax(const float* __restrict__ lg,
                                                    float* __restrict__ out, int m0) {
  int rl = blockIdx.x, tid = threadIdx.x;
  int row = m0 + rl;
  int l = row >> 4, b = row & 15;
  const float* in = lg + (long)rl * 10001;
  float* op = out + ((long)b * 512 + l) * 10001;
  __shared__ float red[256];
  float m = -3.4e38f;
  for (int v = tid; v < 10001; v += 256) m = fmaxf(m, in[v]);
  red[tid] = m; __syncthreads();
  for (int st = 128; st > 0; st >>= 1) { if (tid < st) red[tid] = fmaxf(red[tid], red[tid + st]); __syncthreads(); }
  m = red[0]; __syncthreads();
  float s = 0.f;
  for (int v = tid; v < 10001; v += 256) s += __expf(in[v] - m);
  red[tid] = s; __syncthreads();
  for (int st = 128; st > 0; st >>= 1) { if (tid < st) red[tid] += red[tid + st]; __syncthreads(); }
  float lse = m + __logf(red[0]);
  for (int v = tid; v < 10001; v += 256) op[v] = in[v] - lse;
}

__global__ void k_zero1(float* p) { *p = 0.f; }

// ---------------------------------------------------------------------------
// Host side
// ---------------------------------------------------------------------------
namespace {

struct Arena {
  char* base; size_t off;
  float* alloc(size_t elems) {
    float* p = (float*)(base + off);
    off += (elems * sizeof(float) + 255) & ~(size_t)255;
    return p;
  }
};

struct ResP  { const float *c1w, *c1b, *c2w, *c2b, *g, *b; };
struct AttnP { const float *inw, *inb, *ow, *ob, *l1w, *l1b, *l2w, *l2b, *g1, *b1, *g2, *b2; };
struct EncP  { const float *cw, *cb; ResP res[4]; const float* dw[2]; const float* db[2]; int down; AttnP attn; };
struct DecP  { const float *sa_inw, *sa_inb, *sa_ow, *sa_ob,
                           *ca_inw, *ca_inb, *ca_ow, *ca_ob,
                           *l1w, *l1b, *l2w, *l2b, *g1, *b1, *g2, *b2, *g3, *b3; };

static inline dim3 g1(long n) { return dim3((unsigned)((n + 255) / 256)); }

static void gemm(hipStream_t st, const float* A, const float* Bm, const float* bias,
                 float* C, int M, int N, int K, int lda, int ldb, int ldc,
                 long sA, long sB, long sC, int batch,
                 float alpha, int transB, int transC, int biasMode, int act) {
  dim3 grid((N + TBN - 1) / TBN, (M + TBM - 1) / TBM, batch);
  k_gemm<<<grid, 256, 0, st>>>(A, Bm, bias, C, M, N, K, lda, ldb, ldc,
                               sA, sB, sC, alpha, transB, transC, biasMode, act);
}

// x[B,Cin,T0] -> out[B,256,T0>>down]
static void run_encoder(const float* xin, int Cin, int T0, const EncP& p,
                        float* out, Arena& ws, hipStream_t stream) {
  size_t mark = ws.off;
  int T = T0;
  float* h   = ws.alloc((size_t)16 * 256 * T0);
  float* t1  = ws.alloc((size_t)16 * 256 * T0);
  float* col = ws.alloc((size_t)16 * 1024 * T0);
  float* t2  = ws.alloc((size_t)16 * 128 * T0);

  // h = elu(conv1x1(x))
  gemm(stream, p.cw, xin, p.cb, h, 256, T, Cin, Cin, T, T,
       0, (long)Cin * T, (long)256 * T, 16, 1.f, 0, 0, 2, 2);

  for (int d = 0; d < p.down; d++) {
    for (int r = 0; r < 2; r++) {
      const ResP& rp = p.res[d * 2 + r];
      k_elu<<<g1((long)16 * 256 * T), 256, 0, stream>>>(t1, h, (long)16 * 256 * T);
      k_im2col3<<<g1((long)16 * 256 * 3 * T), 256, 0, stream>>>(t1, col, 256, T, (long)16 * 256 * 3 * T);
      // elu fused (feeds elu(h1) into conv2)
      gemm(stream, rp.c1w, col, rp.c1b, t2, 128, T, 768, 768, T, T,
           0, (long)768 * T, (long)128 * T, 16, 1.f, 0, 0, 2, 2);
      gemm(stream, rp.c2w, t2, rp.c2b, t1, 256, T, 128, 128, T, T,
           0, (long)128 * T, (long)256 * T, 16, 1.f, 0, 0, 2, 0);
      k_cln<<<g1((long)16 * T), 256, 0, stream>>>(t1, h, h, rp.g, rp.b, T, 16 * T);
    }
    int T2 = T >> 1;
    k_im2col4<<<g1((long)16 * 1024 * T2), 256, 0, stream>>>(h, col, 256, T, T2, (long)16 * 1024 * T2);
    gemm(stream, p.dw[d], col, p.db[d], t1, 256, T2, 1024, 1024, T2, T2,
         0, (long)1024 * T2, (long)256 * T2, 16, 1.f, 0, 0, 2, 2);
    float* tmp = h; h = t1; t1 = tmp;
    T = T2;
  }

  // post-norm transformer encoder layer (H=1, D=256, FF=256), skip-add
  int S = T, rows = S * 16;
  float* xT = ws.alloc((size_t)rows * 256);
  k_bct_to_tbc<<<g1((long)16 * 256 * S), 256, 0, stream>>>(h, xT, S, (long)16 * 256 * S);
  float* q  = ws.alloc((size_t)rows * 256);
  float* kb = ws.alloc((size_t)rows * 256);
  float* v  = ws.alloc((size_t)rows * 256);
  float* sc = ws.alloc((size_t)16 * S * S);
  float* o  = ws.alloc((size_t)rows * 256);
  float* ao = ws.alloc((size_t)rows * 256);
  const AttnP& a = p.attn;
  gemm(stream, xT, a.inw,             a.inb,       q,  rows, 256, 256, 256, 256, 256, 0, 0, 0, 1, 1.f, 1, 0, 1, 0);
  gemm(stream, xT, a.inw + 256 * 256, a.inb + 256, kb, rows, 256, 256, 256, 256, 256, 0, 0, 0, 1, 1.f, 1, 0, 1, 0);
  gemm(stream, xT, a.inw + 512 * 256, a.inb + 512, v,  rows, 256, 256, 256, 256, 256, 0, 0, 0, 1, 1.f, 1, 0, 1, 0);
  gemm(stream, q, kb, nullptr, sc, S, S, 256, 16 * 256, 16 * 256, S,
       256, 256, (long)S * S, 16, 0.0625f, 1, 0, 0, 0);
  k_softmax<<<dim3(16 * S), 256, 0, stream>>>(sc, S, S, 0);
  gemm(stream, sc, v, nullptr, o, S, 256, S, S, 16 * 256, 16 * 256,
       (long)S * S, 256, 256, 16, 1.f, 0, 0, 0, 0);
  gemm(stream, o, a.ow, a.ob, ao, rows, 256, 256, 256, 256, 256, 0, 0, 0, 1, 1.f, 1, 0, 1, 0);
  k_ln<<<dim3(rows), 256, 0, stream>>>(xT, xT, ao, a.g1, a.b1);
  gemm(stream, xT, a.l1w, a.l1b, q,  rows, 256, 256, 256, 256, 256, 0, 0, 0, 1, 1.f, 1, 0, 1, 1);
  gemm(stream, q,  a.l2w, a.l2b, ao, rows, 256, 256, 256, 256, 256, 0, 0, 0, 1, 1.f, 1, 0, 1, 0);
  k_ln<<<dim3(rows), 256, 0, stream>>>(xT, xT, ao, a.g2, a.b2);
  k_tbc_add_skip<<<g1((long)16 * 256 * S), 256, 0, stream>>>(xT, h, out, S, (long)16 * 256 * S);

  ws.off = mark;
}

} // namespace

extern "C" void kernel_launch(void* const* d_in, const int* in_sizes, int n_in,
                              void* d_out, int out_size, void* d_ws, size_t ws_size,
                              hipStream_t stream) {
  (void)in_sizes; (void)n_in; (void)out_size; (void)ws_size;

  // -------- input walk (setup_inputs dict order, recursive insertion order) --
  int ii = 0;
  const int* xi = (const int*)d_in[ii++];            // x [16,512]
  auto F = [&]() -> const float* { return (const float*)d_in[ii++]; };
  const float* p_embed = F();                         // [10000,256]
  EncP enc[3];
  auto readEnc = [&](EncP& e, int down) {
    e.down = down;
    e.cw = F(); e.cb = F();
    for (int r = 0; r < down * 2; r++) {
      e.res[r].c1w = F(); e.res[r].c1b = F();
      e.res[r].c2w = F(); e.res[r].c2b = F();
      e.res[r].g = F();   e.res[r].b = F();
    }
    for (int d = 0; d < down; d++) { e.dw[d] = F(); e.db[d] = F(); }
    AttnP& a = e.attn;
    a.inw = F(); a.inb = F(); a.ow = F(); a.ob = F();
    a.l1w = F(); a.l1b = F(); a.l2w = F(); a.l2b = F();
    a.g1 = F();  a.b1 = F();  a.g2 = F();  a.b2 = F();
  };
  readEnc(enc[0], 2); readEnc(enc[1], 2); readEnc(enc[2], 1);
  const float *qcw[3], *qcb[3];
  for (int i = 0; i < 3; i++) { qcw[i] = F(); qcb[i] = F(); }
  const float* vq[3];
  for (int i = 0; i < 3; i++) vq[i] = F();
  const float *cqw[3], *cqb[3];
  for (int i = 0; i < 3; i++) { cqw[i] = F(); cqb[i] = F(); }
  const float* dec_embed = F();                       // [10001,256]
  DecP dl[6];
  for (int i = 0; i < 6; i++) {
    DecP& L = dl[i];
    L.sa_inw = F(); L.sa_inb = F(); L.sa_ow = F(); L.sa_ob = F();
    L.ca_inw = F(); L.ca_inb = F(); L.ca_ow = F(); L.ca_ob = F();
    L.l1w = F(); L.l1b = F(); L.l2w = F(); L.l2b = F();
    L.g1 = F(); L.b1 = F(); L.g2 = F(); L.b2 = F(); L.g3 = F(); L.b3 = F();
  }
  const float* out_w = F();                           // [10001,256]
  const float* out_b = F();                           // [10001]

  // -------- output layout (return-order concat) -----------------------------
  float* dout = (float*)d_out;
  const long off_q[3]  = { 16L * 512 * 10001,
                           16L * 512 * 10001 + 16L * 128 * 64,
                           16L * 512 * 10001 + 16L * 128 * 64 + 16L * 32 * 64 };
  const long off_diff  = off_q[2] + 16L * 16 * 64;
  const long off_i[3]  = { off_diff + 1, off_diff + 1 + 16L * 128,
                           off_diff + 1 + 16L * 128 + 16L * 32 };

  // -------- workspace -------------------------------------------------------
  Arena ws{ (char*)d_ws, 0 };
  float* e1   = ws.alloc(16L * 256 * 128);
  float* e2   = ws.alloc(16L * 256 * 32);
  float* e3   = ws.alloc(16L * 256 * 16);
  float* memb = ws.alloc(176L * 16 * 256);   // [Tm=176, B, 256]
  float* tdec = ws.alloc(8192L * 256);       // [L,B,256]
  size_t markTop = ws.off;

  // -------- encoders --------------------------------------------------------
  float* h0 = ws.alloc(16L * 256 * 512);
  k_embed_enc<<<dim3(16 * 512), 256, 0, stream>>>(xi, p_embed, h0);
  run_encoder(h0, 256, 512, enc[0], e1, ws, stream);
  run_encoder(e1, 256, 128, enc[1], e2, ws, stream);
  run_encoder(e2, 256, 32,  enc[2], e3, ws, stream);
  ws.off = markTop;

  // -------- quantize + memory ----------------------------------------------
  k_zero1<<<1, 1, 0, stream>>>(dout + off_diff);
  {
    float* xq = ws.alloc(16L * 128 * 64);
    const float* eouts[3] = { e1, e2, e3 };
    const int Ts[3] = { 128, 32, 16 };
    const int toffs[3] = { 0, 128, 160 };
    for (int i = 0; i < 3; i++) {
      int Ti = Ts[i];
      // pre-VQ conv1x1 (writes [B,T,64] via transC)
      gemm(stream, qcw[i], eouts[i], qcb[i], xq, 64, Ti, 256, 256, Ti, 64,
           0, (long)256 * Ti, (long)Ti * 64, 16, 1.f, 0, 1, 2, 0);
      k_quantize<<<dim3(16 * Ti), 256, 0, stream>>>(
          xq, vq[i], dout + off_q[i], dout + off_i[i], dout + off_diff,
          1.f / (float)(16 * Ti * 64));
      // post-VQ conv1x1, scatter into mem [Tm,B,256]
      gemm(stream, cqw[i], dout + off_q[i], cqb[i],
           memb + (long)toffs[i] * 16 * 256, 256, Ti, 64, 64, 64, 16 * 256,
           0, (long)Ti * 64, 256, 16, 1.f, 1, 1, 2, 0);
    }
    ws.off = markTop;
  }

  // -------- decoder ---------------------------------------------------------
  k_embed_dec<<<dim3(512 * 16), 256, 0, stream>>>(xi, dec_embed, tdec);
  {
    size_t markDec = ws.off;
    float* q    = ws.alloc(8192L * 256);
    float* kbuf = ws.alloc(8192L * 256);
    float* vbuf = ws.alloc(8192L * 256);
    float* obuf = ws.alloc(8192L * 256);
    float* ao   = ws.alloc(8192L * 256);
    float* sc   = ws.alloc(64L * 512 * 512);
    float* f1   = ws.alloc(8192L * 1024);
    for (int li = 0; li < 6; li++) {
      const DecP& L = dl[li];
      // ---- causal self-attention (H=4, Dh=64; per-(b,h) offset = z*64) ----
      gemm(stream, tdec, L.sa_inw,             L.sa_inb,       q,    8192, 256, 256, 256, 256, 256, 0, 0, 0, 1, 1.f, 1, 0, 1, 0);
      gemm(stream, tdec, L.sa_inw + 256 * 256, L.sa_inb + 256, kbuf, 8192, 256, 256, 256, 256, 256, 0, 0, 0, 1, 1.f, 1, 0, 1, 0);
      gemm(stream, tdec, L.sa_inw + 512 * 256, L.sa_inb + 512, vbuf, 8192, 256, 256, 256, 256, 256, 0, 0, 0, 1, 1.f, 1, 0, 1, 0);
      gemm(stream, q, kbuf, nullptr, sc, 512, 512, 64, 4096, 4096, 512,
           64, 64, 512L * 512, 64, 0.125f, 1, 0, 0, 0);
      k_softmax<<<dim3(64 * 512), 256, 0, stream>>>(sc, 512, 512, 1);
      gemm(stream, sc, vbuf, nullptr, obuf, 512, 64, 512, 512, 4096, 4096,
           512L * 512, 64, 64, 64, 1.f, 0, 0, 0, 0);
      gemm(stream, obuf, L.sa_ow, L.sa_ob, ao, 8192, 256, 256, 256, 256, 256, 0, 0, 0, 1, 1.f, 1, 0, 1, 0);
      k_ln<<<dim3(8192), 256, 0, stream>>>(tdec, tdec, ao, L.g1, L.b1);
      // ---- cross-attention over mem (Tm=176) ----
      gemm(stream, tdec, L.ca_inw,             L.ca_inb,       q,    8192, 256, 256, 256, 256, 256, 0, 0, 0, 1, 1.f, 1, 0, 1, 0);
      gemm(stream, memb, L.ca_inw + 256 * 256, L.ca_inb + 256, kbuf, 2816, 256, 256, 256, 256, 256, 0, 0, 0, 1, 1.f, 1, 0, 1, 0);
      gemm(stream, memb, L.ca_inw + 512 * 256, L.ca_inb + 512, vbuf, 2816, 256, 256, 256, 256, 256, 0, 0, 0, 1, 1.f, 1, 0, 1, 0);
      gemm(stream, q, kbuf, nullptr, sc, 512, 176, 64, 4096, 4096, 176,
           64, 64, 512L * 176, 64, 0.125f, 1, 0, 0, 0);
      k_softmax<<<dim3(64 * 512), 256, 0, stream>>>(sc, 176, 512, 0);
      gemm(stream, sc, vbuf, nullptr, obuf, 512, 64, 176, 176, 4096, 4096,
           512L * 176, 64, 64, 64, 1.f, 0, 0, 0, 0);
      gemm(stream, obuf, L.ca_ow, L.ca_ob, ao, 8192, 256, 256, 256, 256, 256, 0, 0, 0, 1, 1.f, 1, 0, 1, 0);
      k_ln<<<dim3(8192), 256, 0, stream>>>(tdec, tdec, ao, L.g2, L.b2);
      // ---- FFN ----
      gemm(stream, tdec, L.l1w, L.l1b, f1, 8192, 1024, 256, 256, 256, 1024, 0, 0, 0, 1, 1.f, 1, 0, 1, 1);
      gemm(stream, f1,   L.l2w, L.l2b, ao, 8192, 256, 1024, 1024, 256, 256, 0, 0, 0, 1, 1.f, 1, 0, 1, 0);
      k_ln<<<dim3(8192), 256, 0, stream>>>(tdec, tdec, ao, L.g3, L.b3);
    }
    ws.off = markDec;
  }

  // -------- vocab projection + log_softmax (chunked over rows) --------------
  {
    float* logits = ws.alloc(2048L * 10001);
    for (int m0 = 0; m0 < 8192; m0 += 2048) {
      gemm(stream, tdec + (long)m0 * 256, out_w, out_b, logits,
           2048, 10001, 256, 256, 256, 10001, 0, 0, 0, 1, 1.f, 1, 0, 1, 0);
      k_logsoftmax<<<dim3(2048), 256, 0, stream>>>(logits, dout, m0);
    }
  }
}
